// ProbSparse_Self_Attention_Block_67654324846597
// MI455X (gfx1250) — matrix operand order repaired and checked by reading, not compiled, not run on hardware
//
#include <hip/hip_runtime.h>
#include <stdint.h>

typedef uint16_t u16;
typedef __attribute__((ext_vector_type(16))) __bf16 v16bf;
typedef __attribute__((ext_vector_type(8)))  float  v8f;

#define B_  4
#define L_  2048
#define D_  512
#define H_  8
#define QD_ 64
#define FF_ 2048
#define M_  (B_ * L_)          // 8192 rows
#define SCALE_ 0.125f          // 1/sqrt(64), exact power of two

struct B32 { uint4 a, b; };

__device__ __forceinline__ u16 f2bf(float f) {
  uint32_t u = __builtin_bit_cast(uint32_t, f);
  u += 0x7FFFu + ((u >> 16) & 1u);     // round-to-nearest-even
  return (u16)(u >> 16);
}

// Build a 16-element bf16 fragment from two 16-byte chunks.
__device__ __forceinline__ v16bf frag16(const u16* p0, const u16* p1) {
  B32 t;
  t.a = *(const uint4*)p0;
  t.b = *(const uint4*)p1;
  return __builtin_bit_cast(v16bf, t);
}

// All-ones bf16 fragment (0x3F80 per half).
__device__ __forceinline__ v16bf ones16() {
  B32 t;
  t.a.x = 0x3F803F80u; t.a.y = 0x3F803F80u; t.a.z = 0x3F803F80u; t.a.w = 0x3F803F80u;
  t.b = t.a;
  return __builtin_bit_cast(v16bf, t);
}

__device__ __forceinline__ v8f wmma_bf16(v16bf a, v16bf b, v8f c) {
  return __builtin_amdgcn_wmma_f32_16x16x32_bf16(
      /*neg_a=*/false, a, /*neg_b=*/false, b,
      /*c_mod=*/(short)0, c, /*reuse_a=*/false, /*reuse_b=*/false);
}

// Max-reduce across each 16-lane group (VALU permlane path, no LDS pipe).
#if __has_builtin(__builtin_amdgcn_permlane16)
__device__ __forceinline__ float pl16x(float v, unsigned s0, unsigned s1) {
  unsigned u = __builtin_bit_cast(unsigned, v);
  unsigned p = __builtin_amdgcn_permlane16(u, u, s0, s1, false, false);
  return __builtin_bit_cast(float, p);
}
__device__ __forceinline__ float grp16_max(float v) {
  v = fmaxf(v, pl16x(v, 0x67452301u, 0xEFCDAB89u));   // xor 1
  v = fmaxf(v, pl16x(v, 0x45670123u, 0xCDEF89ABu));   // xor 2
  v = fmaxf(v, pl16x(v, 0x01234567u, 0x89ABCDEFu));   // xor 4
  v = fmaxf(v, pl16x(v, 0xFEDCBA98u, 0x76543210u));   // xor 8
  return v;
}
#else
__device__ __forceinline__ float grp16_max(float v) {
  for (int off = 8; off; off >>= 1) v = fmaxf(v, __shfl_xor(v, off, 32));
  return v;
}
#endif

// ---------------------------------------------------------------------------
// Elementwise conversions
// ---------------------------------------------------------------------------
__global__ __launch_bounds__(256) void f32_to_bf16_kernel(
    const float* __restrict__ in, u16* __restrict__ out, int n) {
  int i = blockIdx.x * 256 + threadIdx.x;
  if (i < n) out[i] = f2bf(in[i]);
}

// out[n*K + k] = in[k*N + n]  (store weight as [N][K] bf16 = column-major B)
__global__ __launch_bounds__(256) void transpose_bf16_kernel(
    const float* __restrict__ in, u16* __restrict__ out, int K, int N) {
  int i = blockIdx.x * 256 + threadIdx.x;
  if (i < K * N) {
    int n = i / K, k = i % K;
    out[i] = f2bf(in[(size_t)k * N + n]);
  }
}

// Vt[((b*H+h)*64 + d)*L + s] = qkv[(b*L+s)*1536 + 1024 + h*64 + d]
__global__ __launch_bounds__(256) void make_vt_kernel(
    const u16* __restrict__ qkv, u16* __restrict__ vt) {
  int i = blockIdx.x * 256 + threadIdx.x;          // 4*8*64*2048 = 2^22 exact
  int s = i & 2047;
  int d = (i >> 11) & 63;
  int h = (i >> 17) & 7;
  int b = i >> 20;
  vt[i] = qkv[((size_t)(b * L_ + s)) * 1536 + 1024 + h * 64 + d];
}

// ---------------------------------------------------------------------------
// WMMA GEMM: C[M,N] = A[M,K](bf16,row-major) x Bt[N,K](bf16)
// One wave computes a 16x64 tile.
// EPI: 0 = f32 out, 1 = bf16 out, 2 = bias+relu -> bf16,
//      3 = bf16 out with Q columns (col<512) pre-scaled by 1/sqrt(QD)
// ---------------------------------------------------------------------------
template <int EPI>
__global__ __launch_bounds__(256) void gemm_bf16_kernel(
    const u16* __restrict__ A, const u16* __restrict__ Bt,
    const float* __restrict__ bias, float* __restrict__ Cf,
    u16* __restrict__ Cb, int N, int K) {
  const int wave = threadIdx.x >> 5;
  const int lane = threadIdx.x & 31;
  const int mrow = lane & 15;         // N index for B/C frag, M row for A frag
  const int g    = lane >> 4;         // half-wave selector
  const int m0   = blockIdx.x * 16;
  const int n0   = (blockIdx.y * 8 + wave) * 64;
  if (n0 >= N) return;

  v8f acc[4] = {};
  const u16* arow = A + (size_t)(m0 + mrow) * K;

  for (int k0 = 0; k0 < K; k0 += 32) {
    // A frag: lane row = mrow, 16 contiguous K at k0 + g*16
    v16bf af = frag16(arow + k0 + g * 16, arow + k0 + g * 16 + 8);
#pragma unroll
    for (int t = 0; t < 4; ++t) {
      // B frag: lane col = mrow, 16 contiguous K at k0 + g*16
      const u16* bp = Bt + (size_t)(n0 + t * 16 + mrow) * K + k0 + g * 16;
      v16bf bf = frag16(bp, bp + 8);
      acc[t] = wmma_bf16(af, bf, acc[t]);
    }
  }

#pragma unroll
  for (int t = 0; t < 4; ++t) {
    const int col = n0 + t * 16 + mrow;
    float bv = (EPI == 2) ? bias[col] : 0.0f;
#pragma unroll
    for (int r = 0; r < 8; ++r) {
      const int row = m0 + r + 8 * g;   // C frag: vgpr r, half g -> M = r + 8g
      float v = acc[t][r];
      if (EPI == 0) {
        Cf[(size_t)row * N + col] = v;
      } else if (EPI == 2) {
        v = fmaxf(v + bv, 0.0f);
        Cb[(size_t)row * N + col] = f2bf(v);
      } else {
        if (EPI == 3 && col < 512) v *= SCALE_;   // fold softmax scale into Q
        Cb[(size_t)row * N + col] = f2bf(v);
      }
    }
  }
}

// ---------------------------------------------------------------------------
// Flash attention: one wave per (b, h, 16-query tile). Online softmax,
// 32 keys per step; scores, PV, and row-sums all via WMMA.
// qkv: [M,1536] bf16 (Q*scale|K|V), Vt: [b,h,64,L] bf16, Zb out: [M,512] bf16.
// ---------------------------------------------------------------------------
__global__ __launch_bounds__(256) void attn_kernel(
    const u16* __restrict__ qkv, const u16* __restrict__ Vt,
    u16* __restrict__ Zb) {
  __shared__ u16 P[8][16][32];        // per-wave probability tile (16q x 32k)

  const int wave = threadIdx.x >> 5;
  const int lane = threadIdx.x & 31;
  const int wid  = blockIdx.x * 8 + wave;   // 0..4095
  const int qt = wid & 127;                 // query tile (L/16 = 128)
  const int h  = (wid >> 7) & 7;
  const int b  = wid >> 10;
  const int mrow = lane & 15;
  const int g    = lane >> 4;
  const int q0   = qt * 16;

  // Q fragments (qd 0..31 and 32..63), row = b*L + q0 + mrow
  const size_t qrow = ((size_t)(b * L_) + q0 + mrow) * 1536 + h * 64;
  v16bf aQ0 = frag16(qkv + qrow + g * 16,      qkv + qrow + g * 16 + 8);
  v16bf aQ1 = frag16(qkv + qrow + 32 + g * 16, qkv + qrow + 32 + g * 16 + 8);
  const v16bf vones = ones16();

  v8f o0 = {}, o1 = {}, o2 = {}, o3 = {}, o4 = {};   // o4 = running row sums
  float m8[8];
#pragma unroll
  for (int r = 0; r < 8; ++r) m8[r] = -INFINITY;

  const u16* vbase = Vt + ((size_t)(b * H_ + h) * 64) * L_;

  for (int s0 = 0; s0 < L_; s0 += 32) {
    // ---- scores: S0 = keys s0..s0+15, S1 = keys s0+16..s0+31 (K = qd = 64)
    v8f S0 = {}, S1 = {};
    {
      size_t kr0 = ((size_t)(b * L_) + s0 + mrow) * 1536 + 512 + h * 64;
      size_t kr1 = kr0 + 16 * 1536;
      v16bf bK;
      bK = frag16(qkv + kr0 + g * 16,      qkv + kr0 + g * 16 + 8);      S0 = wmma_bf16(aQ0, bK, S0);
      bK = frag16(qkv + kr0 + 32 + g * 16, qkv + kr0 + 32 + g * 16 + 8); S0 = wmma_bf16(aQ1, bK, S0);
      bK = frag16(qkv + kr1 + g * 16,      qkv + kr1 + g * 16 + 8);      S1 = wmma_bf16(aQ0, bK, S1);
      bK = frag16(qkv + kr1 + 32 + g * 16, qkv + kr1 + 32 + g * 16 + 8); S1 = wmma_bf16(aQ1, bK, S1);
    }

    // ---- online softmax per row (row = r + 8g lives in vgpr r, 16 lanes)
#pragma unroll
    for (int r = 0; r < 8; ++r) {
      float x0 = S0[r];                    // already scaled (Q pre-scaled)
      float x1 = S1[r];
      float nm = fmaxf(m8[r], grp16_max(fmaxf(x0, x1)));
      float al = __expf(m8[r] - nm);
      m8[r] = nm;
      o0[r] *= al; o1[r] *= al; o2[r] *= al; o3[r] *= al; o4[r] *= al;
      const int prow = r + 8 * g;
      P[wave][prow][mrow]      = f2bf(__expf(x0 - nm));
      P[wave][prow][16 + mrow] = f2bf(__expf(x1 - nm));
    }

    // ---- P (16x32) as A fragment from LDS
    v16bf aP = frag16(&P[wave][mrow][g * 16], &P[wave][mrow][g * 16 + 8]);

    // ---- row sums via WMMA: o4 += P x ones  (every column = row sum)
    o4 = wmma_bf16(aP, vones, o4);

    // ---- P x V: 4 output qd tiles; Vt gives contiguous 16 keys per lane
    {
      const u16* vp;
      vp = vbase + (size_t)(0 * 16 + mrow) * L_ + s0 + g * 16; o0 = wmma_bf16(aP, frag16(vp, vp + 8), o0);
      vp = vbase + (size_t)(1 * 16 + mrow) * L_ + s0 + g * 16; o1 = wmma_bf16(aP, frag16(vp, vp + 8), o1);
      vp = vbase + (size_t)(2 * 16 + mrow) * L_ + s0 + g * 16; o2 = wmma_bf16(aP, frag16(vp, vp + 8), o2);
      vp = vbase + (size_t)(3 * 16 + mrow) * L_ + s0 + g * 16; o3 = wmma_bf16(aP, frag16(vp, vp + 8), o3);
    }
  }

  // ---- normalize by row sums (o4) and write Z (bf16)
#pragma unroll
  for (int r = 0; r < 8; ++r) {
    const int row = r + 8 * g;
    const float inv = 1.0f / o4[r];
    const size_t ob = ((size_t)(b * L_) + q0 + row) * 512 + h * 64 + mrow;
    Zb[ob +  0] = f2bf(o0[r] * inv);
    Zb[ob + 16] = f2bf(o1[r] * inv);
    Zb[ob + 32] = f2bf(o2[r] * inv);
    Zb[ob + 48] = f2bf(o3[r] * inv);
  }
}

// ---------------------------------------------------------------------------
// LayerNorm over D=512: y = LN(in + bias + resid)*gamma + beta
// One block (256 threads) per row; optional bf16 copy.
// ---------------------------------------------------------------------------
__global__ __launch_bounds__(256) void ln_kernel(
    const float* __restrict__ in, const float* __restrict__ bias,
    const float* __restrict__ resid, const float* __restrict__ gamma,
    const float* __restrict__ beta, float* __restrict__ outf,
    u16* __restrict__ outb) {
  const int row = blockIdx.x, t = threadIdx.x;
  const size_t base = (size_t)row * 512;
  float v0 = in[base + t]       + bias[t]       + resid[base + t];
  float v1 = in[base + 256 + t] + bias[256 + t] + resid[base + 256 + t];

  __shared__ float red[8];
  float s = v0 + v1;
#pragma unroll
  for (int off = 16; off; off >>= 1) s += __shfl_xor(s, off, 32);
  if ((t & 31) == 0) red[t >> 5] = s;
  __syncthreads();
  float mu = 0.0f;
#pragma unroll
  for (int i = 0; i < 8; ++i) mu += red[i];
  mu *= (1.0f / 512.0f);
  __syncthreads();

  float d0 = v0 - mu, d1 = v1 - mu;
  float q = d0 * d0 + d1 * d1;
#pragma unroll
  for (int off = 16; off; off >>= 1) q += __shfl_xor(q, off, 32);
  if ((t & 31) == 0) red[t >> 5] = q;
  __syncthreads();
  float var = 0.0f;
#pragma unroll
  for (int i = 0; i < 8; ++i) var += red[i];
  var *= (1.0f / 512.0f);

  const float rs = rsqrtf(var + 1e-5f);
  float y0 = d0 * rs * gamma[t]       + beta[t];
  float y1 = d1 * rs * gamma[256 + t] + beta[256 + t];
  outf[base + t]       = y0;
  outf[base + 256 + t] = y1;
  if (outb) {
    outb[base + t]       = f2bf(y0);
    outb[base + 256 + t] = f2bf(y1);
  }
}

// ---------------------------------------------------------------------------
extern "C" void kernel_launch(void* const* d_in, const int* in_sizes, int n_in,
                              void* d_out, int out_size, void* d_ws,
                              size_t ws_size, hipStream_t stream) {
  (void)in_sizes; (void)n_in; (void)out_size; (void)ws_size;
  const float* x     = (const float*)d_in[0];
  const float* WQ    = (const float*)d_in[1];
  const float* WK    = (const float*)d_in[2];
  const float* WV    = (const float*)d_in[3];
  const float* WZ    = (const float*)d_in[4];
  const float* bZ    = (const float*)d_in[5];
  const float* W1    = (const float*)d_in[6];
  const float* b1    = (const float*)d_in[7];
  const float* W2    = (const float*)d_in[8];
  const float* b2    = (const float*)d_in[9];
  const float* gamma = (const float*)d_in[10];
  const float* beta  = (const float*)d_in[11];
  float* out = (float*)d_out;

  char* ws = (char*)d_ws;
  size_t off = 0;
  auto alloc = [&](size_t bytes) -> void* {
    void* p = ws + off;
    off = (off + bytes + 255) & ~(size_t)255;
    return p;
  };
  u16* xb    = (u16*)alloc((size_t)M_ * 512 * 2);
  u16* Wqkvt = (u16*)alloc((size_t)1536 * 512 * 2);
  u16* WZt   = (u16*)alloc((size_t)512 * 512 * 2);
  u16* W1t   = (u16*)alloc((size_t)2048 * 512 * 2);
  u16* W2t   = (u16*)alloc((size_t)512 * 2048 * 2);
  u16* qkv   = (u16*)alloc((size_t)M_ * 1536 * 2);
  u16* Vt    = (u16*)alloc((size_t)B_ * H_ * 64 * L_ * 2);
  u16* zattn = (u16*)alloc((size_t)M_ * 512 * 2);
  float* tmpf = (float*)alloc((size_t)M_ * 512 * 4);
  float* zf   = (float*)alloc((size_t)M_ * 512 * 4);
  u16* zb    = (u16*)alloc((size_t)M_ * 512 * 2);
  u16* hb    = (u16*)alloc((size_t)M_ * 2048 * 2);

  const dim3 blk(256);
  // x -> bf16
  f32_to_bf16_kernel<<<(M_ * 512 + 255) / 256, blk, 0, stream>>>(x, xb, M_ * 512);
  // weights -> bf16, transposed to [N][K]
  transpose_bf16_kernel<<<(512 * 512 + 255) / 256, blk, 0, stream>>>(WQ, Wqkvt, 512, 512);
  transpose_bf16_kernel<<<(512 * 512 + 255) / 256, blk, 0, stream>>>(WK, Wqkvt + 512 * 512, 512, 512);
  transpose_bf16_kernel<<<(512 * 512 + 255) / 256, blk, 0, stream>>>(WV, Wqkvt + 1024 * 512, 512, 512);
  transpose_bf16_kernel<<<(512 * 512 + 255) / 256, blk, 0, stream>>>(WZ, WZt, 512, 512);
  transpose_bf16_kernel<<<(512 * 2048 + 255) / 256, blk, 0, stream>>>(W1, W1t, 512, 2048);
  transpose_bf16_kernel<<<(2048 * 512 + 255) / 256, blk, 0, stream>>>(W2, W2t, 2048, 512);

  // fused QKV projection: [8192,512] x [512,1536] -> bf16 qkv (Q pre-scaled)
  gemm_bf16_kernel<3><<<dim3(M_ / 16, 3), blk, 0, stream>>>(
      xb, Wqkvt, nullptr, nullptr, qkv, 1536, 512);
  // V transpose for PV fragments
  make_vt_kernel<<<(B_ * H_ * 64 * L_) / 256, blk, 0, stream>>>(qkv, Vt);
  // flash attention -> zattn bf16
  attn_kernel<<<512, blk, 0, stream>>>(qkv, Vt, zattn);
  // Z projection (f32 raw)
  gemm_bf16_kernel<0><<<dim3(M_ / 16, 1), blk, 0, stream>>>(
      zattn, WZt, nullptr, tmpf, nullptr, 512, 512);
  // LN1: z = LN(zattn@WZ + bZ + x)
  ln_kernel<<<M_, blk, 0, stream>>>(tmpf, bZ, x, gamma, beta, zf, zb);
  // FFN1 with bias+ReLU epilogue -> bf16 hidden
  gemm_bf16_kernel<2><<<dim3(M_ / 16, 4), blk, 0, stream>>>(
      zb, W1t, b1, nullptr, hb, 2048, 512);
  // FFN2 (f32 raw)
  gemm_bf16_kernel<0><<<dim3(M_ / 16, 1), blk, 0, stream>>>(
      hb, W2t, nullptr, tmpf, nullptr, 512, 2048);
  // LN2: out = LN(hdn@W2 + b2 + z)
  ln_kernel<<<M_, blk, 0, stream>>>(tmpf, b2, zf, gamma, beta, out, nullptr);
}
